// Mlp_2_Layer_MMoE_73048803770373
// MI455X (gfx1250) — compile-verified, matrix operand order checked by reading, not run on hardware
//
#include <hip/hip_runtime.h>
#include <hip/hip_bf16.h>
#include <math.h>

typedef __attribute__((ext_vector_type(16))) __bf16 v16bf;
typedef __attribute__((ext_vector_type(8)))  float  v8f;

#define B_SZ 131072
#define IN_F 113
#define N_E  4
#define N_D  8
#define H1_F 256
#define H2_F 113
#define H2_P 128      // task padded to 128 cols
#define T1_F 256
#define T2_F 128

// =====================================================================
// Packing kernels: put B (weights) and emb-A into WMMA fragment order,
// bf16, zero-padded. Fragment tile = 32 lanes x 16 bf16 = 1 KB.
// Element j of lane L -> K = kt*32 + 16*(j>>3) + 8*(L>>4) + (j&7)
//                        N/M = base + (L&15)
// =====================================================================
__global__ void __launch_bounds__(256)
pack_w(const float* __restrict__ W, __bf16* __restrict__ Wp,
       int K, int N, int KT, int NT, int G)
{
  int idx = blockIdx.x * 256 + threadIdx.x;     // ((g*KT+kt)*NT+nt)*32 + lane
  int total = G * KT * NT * 32;
  if (idx >= total) return;
  int lane = idx & 31;
  int t = idx >> 5;
  int nt = t % NT; t /= NT;
  int kt = t % KT; int g = t / KT;
  int n  = nt * 16 + (lane & 15);
  int hi = (lane >> 4) & 1;
  const float* Wg = W + (size_t)g * K * N;
  __bf16* outp = Wp + (size_t)idx * 16;
#pragma unroll
  for (int j = 0; j < 16; ++j) {
    int k = kt * 32 + ((j >> 3) << 4) + (hi << 3) + (j & 7);
    float x = (k < K && n < N) ? Wg[(size_t)k * N + n] : 0.0f;
    outp[j] = (__bf16)x;
  }
}

__global__ void __launch_bounds__(256)
pack_emb(const float* __restrict__ emb, __bf16* __restrict__ Ap)
{
  int idx = blockIdx.x * 256 + threadIdx.x;     // (mt*4 + kt)*32 + lane
  int lane = idx & 31;
  int t = idx >> 5;
  int kt = t & 3;
  int mt = t >> 2;
  if (mt >= B_SZ / 16) return;
  int m  = mt * 16 + (lane & 15);
  int hi = (lane >> 4) & 1;
  const float* row = emb + (size_t)m * IN_F;
  __bf16* outp = Ap + (size_t)idx * 16;
#pragma unroll
  for (int j = 0; j < 16; ++j) {
    int k = kt * 32 + ((j >> 3) << 4) + (hi << 3) + (j & 7);
    outp[j] = (__bf16)((k < IN_F) ? row[k] : 0.0f);
  }
}

// =====================================================================
// Fully-static grouped GEMM:  Z[b,g,n] = act(A)[b,g,:] @ W[g] + bias[g]
//  - K multiple of 32 for non-packed A (256 or 128); no inner guards
//  - B from packed bf16 fragments (zero-padded), batch-preloaded per k-step
//  - AFFINE: fold y = relu(x*sc[k]+sh[k]) into the A-fragment load
//  - Fused BN statistics: per-feature sum/sum^2 reduced in LDS across the
//    block's 8 waves, then one global atomicAdd per column per block
//  - 8 waves/block; wave owns 16 rows x NTILES*16 cols
// =====================================================================
template<int K, int NREAL, int AFFINE, int PACKED_A, int NTILES>
__global__ void __launch_bounds__(256)
gemm_t(const float* __restrict__ A, size_t a_row_stride, size_t a_g_stride,
       const __bf16* __restrict__ Ap,
       const __bf16* __restrict__ Wp,
       const float* __restrict__ bias,
       const float* __restrict__ sc, const float* __restrict__ sh,
       float* __restrict__ Z, float* __restrict__ sums, int G)
{
  constexpr int KT = (K + 31) / 32;
  constexpr int NT = (NREAL + 15) / 16;

  const int lane = threadIdx.x & 31;
  const int wave = threadIdx.x >> 5;
  const int g    = blockIdx.z;
  const int m0   = blockIdx.y * 128 + wave * 16;
  const int nt0  = blockIdx.x * NTILES;

  __shared__ float lsum[NTILES * 16];
  __shared__ float lsq [NTILES * 16];
  if (threadIdx.x < NTILES * 16) {
    lsum[threadIdx.x] = 0.0f;
    lsq [threadIdx.x] = 0.0f;
  }

  const __bf16* WpG = Wp + ((size_t)g * KT * NT) * 32 * 16;

  const float*  row = nullptr;
  const __bf16* apg = nullptr;
  if constexpr (PACKED_A) {
    apg = Ap + ((size_t)(m0 >> 4) * KT * 32 + lane) * 16;
  } else {
    row = A + (size_t)g * a_g_stride
            + (size_t)(m0 + (lane & 15)) * a_row_stride
            + ((lane >> 4) << 3);
  }
  const float* scp = sc + (size_t)g * K + ((lane >> 4) << 3);
  const float* shp = sh + (size_t)g * K + ((lane >> 4) << 3);

  v8f acc[NTILES];
#pragma unroll
  for (int t = 0; t < NTILES; ++t) acc[t] = v8f{};

#pragma unroll
  for (int kt = 0; kt < KT; ++kt) {
    v16bf a;
    if constexpr (PACKED_A) {
      a = *(const v16bf*)(apg + (size_t)kt * 32 * 16);
    } else {
      const float* rp = row + kt * 32;
      float4 x0 = *(const float4*)(rp);
      float4 x1 = *(const float4*)(rp + 4);
      float4 x2 = *(const float4*)(rp + 16);
      float4 x3 = *(const float4*)(rp + 20);
      float vv[16] = {x0.x, x0.y, x0.z, x0.w, x1.x, x1.y, x1.z, x1.w,
                      x2.x, x2.y, x2.z, x2.w, x3.x, x3.y, x3.z, x3.w};
      if constexpr (AFFINE) {
        const float* sr = scp + kt * 32;
        const float* hr = shp + kt * 32;
        float4 s0 = *(const float4*)(sr);
        float4 s1 = *(const float4*)(sr + 4);
        float4 s2 = *(const float4*)(sr + 16);
        float4 s3 = *(const float4*)(sr + 20);
        float4 h0 = *(const float4*)(hr);
        float4 h1 = *(const float4*)(hr + 4);
        float4 h2 = *(const float4*)(hr + 16);
        float4 h3 = *(const float4*)(hr + 20);
        float ss[16] = {s0.x, s0.y, s0.z, s0.w, s1.x, s1.y, s1.z, s1.w,
                        s2.x, s2.y, s2.z, s2.w, s3.x, s3.y, s3.z, s3.w};
        float hh[16] = {h0.x, h0.y, h0.z, h0.w, h1.x, h1.y, h1.z, h1.w,
                        h2.x, h2.y, h2.z, h2.w, h3.x, h3.y, h3.z, h3.w};
#pragma unroll
        for (int j = 0; j < 16; ++j)
          vv[j] = fmaxf(fmaf(vv[j], ss[j], hh[j]), 0.0f);
      }
#pragma unroll
      for (int j = 0; j < 16; ++j) a[j] = (__bf16)vv[j];
    }

    // Batch-preload all B fragments for this k-step, then run the WMMA chain
    const __bf16* wt = WpG + (((size_t)kt * NT + nt0) * 32 + lane) * 16;
    v16bf bfr[NTILES];
#pragma unroll
    for (int t = 0; t < NTILES; ++t)
      bfr[t] = *(const v16bf*)(wt + (size_t)t * 32 * 16);
#pragma unroll
    for (int t = 0; t < NTILES; ++t)
      acc[t] = __builtin_amdgcn_wmma_f32_16x16x32_bf16(
          false, a, false, bfr[t], (short)0, acc[t], false, false);
  }

  // Epilogue: store + fused BN statistics.
  // D layout per ISA (VGPR r -> M = r + 8*(lane>>4), N = lane&15)
  __syncthreads();                 // lsum/lsq zero-init visible
  const float* bg   = bias + (size_t)g * NREAL;
  const int    mrow = m0 + ((lane >> 4) << 3);
  const int    ncol = nt0 * 16 + (lane & 15);
#pragma unroll
  for (int t = 0; t < NTILES; ++t) {
    const int n = ncol + t * 16;
    if ((NREAL % 16 == 0) || (n < NREAL)) {
      const float bv = bg[n];
      float s = 0.0f, q = 0.0f;
#pragma unroll
      for (int r = 0; r < 8; ++r) {
        float v = acc[t][r] + bv;
        Z[((size_t)(mrow + r) * G + g) * NREAL + n] = v;
        s += v;
        q += v * v;
      }
      atomicAdd(&lsum[t * 16 + (lane & 15)], s);
      atomicAdd(&lsq [t * 16 + (lane & 15)], q);
    }
  }
  __syncthreads();
  if (threadIdx.x < NTILES * 16) {
    const int n = nt0 * 16 + threadIdx.x;
    if ((NREAL % 16 == 0) || (n < NREAL)) {
      const int F = G * NREAL;
      atomicAdd(&sums[(size_t)g * NREAL + n],     lsum[threadIdx.x]);
      atomicAdd(&sums[F + (size_t)g * NREAL + n], lsq [threadIdx.x]);
    }
  }
}

// ---------------- BN stats finalize ----------------
__global__ void zero_f32(float* __restrict__ p, int n)
{
  int i = blockIdx.x * 256 + threadIdx.x;
  if (i < n) p[i] = 0.0f;
}

__global__ void bn_finalize(const float* __restrict__ sums,
                            const float* __restrict__ gamma, const float* __restrict__ beta,
                            float* __restrict__ sc, float* __restrict__ sh, int F)
{
  int i = blockIdx.x * 256 + threadIdx.x;
  if (i >= F) return;
  const float invB = 1.0f / (float)B_SZ;
  float mean = sums[i] * invB;
  float var  = sums[F + i] * invB - mean * mean;
  float s = gamma[i] * rsqrtf(var + 1e-5f);
  sc[i] = s;
  sh[i] = beta[i] - mean * s;
}

// ---------------- Gates: softmax(emb @ Wg[d] + bg[d]) over E=4 ----------------
__global__ void __launch_bounds__(256)
gates_kernel(const float* __restrict__ emb, const float* __restrict__ Wg,
             const float* __restrict__ bg, float* __restrict__ gates)
{
  int idx = blockIdx.x * 256 + threadIdx.x;      // idx = b*D + d
  if (idx >= B_SZ * N_D) return;
  int d = idx & (N_D - 1);
  int b = idx >> 3;
  const float* e = emb + (size_t)b * IN_F;
  const float* w = Wg + (size_t)d * IN_F * N_E;
  float l0 = bg[d * N_E + 0], l1 = bg[d * N_E + 1];
  float l2 = bg[d * N_E + 2], l3 = bg[d * N_E + 3];
  for (int k = 0; k < IN_F; ++k) {
    float x = e[k];
    l0 += x * w[k * 4 + 0]; l1 += x * w[k * 4 + 1];
    l2 += x * w[k * 4 + 2]; l3 += x * w[k * 4 + 3];
  }
  float mx = fmaxf(fmaxf(l0, l1), fmaxf(l2, l3));
  float e0 = expf(l0 - mx), e1 = expf(l1 - mx), e2 = expf(l2 - mx), e3 = expf(l3 - mx);
  float inv = 1.0f / (e0 + e1 + e2 + e3);
  float* gp = gates + (size_t)idx * 4;
  gp[0] = e0 * inv; gp[1] = e1 * inv; gp[2] = e2 * inv; gp[3] = e3 * inv;
}

// ------ Mixture: task[b,d,c] = sum_e gates[b,d,e]*relu(bn(z2[b,e,c])), padded to 128 ----
__global__ void __launch_bounds__(128)
task_kernel(const float* __restrict__ z2,
            const float* __restrict__ sc2, const float* __restrict__ sh2,
            const float* __restrict__ gates, float* __restrict__ task)
{
  __shared__ float gs[N_D * N_E];
  const int b = blockIdx.x;
  const int c = threadIdx.x;
  if (c < N_D * N_E) gs[c] = gates[(size_t)b * N_D * N_E + c];
  __syncthreads();
  float fea[N_E];
#pragma unroll
  for (int e = 0; e < N_E; ++e) {
    float f = 0.0f;
    if (c < H2_F) {
      float x = z2[((size_t)b * N_E + e) * H2_F + c];
      f = fmaxf(x * sc2[e * H2_F + c] + sh2[e * H2_F + c], 0.0f);
    }
    fea[e] = f;
  }
#pragma unroll
  for (int d = 0; d < N_D; ++d) {
    float acc = 0.0f;
#pragma unroll
    for (int e = 0; e < N_E; ++e) acc += gs[d * N_E + e] * fea[e];
    task[((size_t)b * N_D + d) * H2_P + c] = acc;   // cols 113..127 get zeros
  }
}

// -------- Final: tower head + sigmoid + domain gather (wave per row) --------
__global__ void __launch_bounds__(256)
final_kernel(const float* __restrict__ z4,
             const float* __restrict__ sc4, const float* __restrict__ sh4,
             const float* __restrict__ Wt3, const float* __restrict__ bt3,
             const int* __restrict__ dom, float* __restrict__ out)
{
  const int lane = threadIdx.x & 31;
  const int wave = threadIdx.x >> 5;
  const int b = blockIdx.x * 8 + wave;
  const int d = dom[b] & (N_D - 1);
  const float* zrow = z4 + ((size_t)b * N_D + d) * T2_F;
  const float* sc = sc4 + (size_t)d * T2_F;
  const float* sh = sh4 + (size_t)d * T2_F;
  const float* w  = Wt3 + (size_t)d * T2_F;
  float acc = 0.0f;
#pragma unroll
  for (int i = 0; i < 4; ++i) {
    int h = lane + 32 * i;
    float t = fmaxf(zrow[h] * sc[h] + sh[h], 0.0f);
    acc += t * w[h];
  }
  acc += __shfl_down(acc, 16, 32);
  acc += __shfl_down(acc,  8, 32);
  acc += __shfl_down(acc,  4, 32);
  acc += __shfl_down(acc,  2, 32);
  acc += __shfl_down(acc,  1, 32);
  if (lane == 0) out[b] = 1.0f / (1.0f + expf(-(acc + bt3[d])));
}

// ---------------- Host-side orchestration ----------------
extern "C" void kernel_launch(void* const* d_in, const int* in_sizes, int n_in,
                              void* d_out, int out_size, void* d_ws, size_t ws_size,
                              hipStream_t stream)
{
  const float* emb = (const float*)d_in[0];
  const int*   dom = (const int*)d_in[1];
  const float* We1 = (const float*)d_in[2];
  const float* be1 = (const float*)d_in[3];
  const float* eg1 = (const float*)d_in[4];
  const float* eb1 = (const float*)d_in[5];
  const float* We2 = (const float*)d_in[6];
  const float* be2 = (const float*)d_in[7];
  const float* eg2 = (const float*)d_in[8];
  const float* eb2 = (const float*)d_in[9];
  const float* Wgp = (const float*)d_in[10];
  const float* bgp = (const float*)d_in[11];
  const float* Wt1 = (const float*)d_in[12];
  const float* bt1 = (const float*)d_in[13];
  const float* tg1 = (const float*)d_in[14];
  const float* tb1 = (const float*)d_in[15];
  const float* Wt2 = (const float*)d_in[16];
  const float* bt2 = (const float*)d_in[17];
  const float* tg2 = (const float*)d_in[18];
  const float* tb2 = (const float*)d_in[19];
  const float* Wt3 = (const float*)d_in[20];
  const float* bt3 = (const float*)d_in[21];
  float* out = (float*)d_out;

  char* ws = (char*)d_ws;
  size_t off = 0;
  auto alloc = [&](size_t bytes) { size_t o = off; off += (bytes + 255) & ~(size_t)255; return o; };

  // Big activation regions (z1/z3 share, z2/z4 share)
  size_t R1  = alloc((size_t)B_SZ * N_D * T1_F * sizeof(float));   // z1 [B,E,256] -> z3 [B,D,256]
  size_t R2  = alloc((size_t)B_SZ * N_D * T2_F * sizeof(float));   // z2 [B,E,113] -> z4 [B,D,128]
  size_t TK  = alloc((size_t)B_SZ * N_D * H2_P * sizeof(float));   // task [B,D,128] padded
  size_t GT  = alloc((size_t)B_SZ * N_D * N_E * sizeof(float));    // gates
  // Packed bf16 operands
  size_t AP  = alloc((size_t)(B_SZ / 16) * 4 * 32 * 16 * 2);       // packed emb (KT=4)
  size_t WP1 = alloc((size_t)N_E * 4 * 16 * 32 * 16 * 2);          // We1: KT=4, NT=16
  size_t WP2 = alloc((size_t)N_E * 8 *  8 * 32 * 16 * 2);          // We2: KT=8, NT=8
  size_t WP3 = alloc((size_t)N_D * 4 * 16 * 32 * 16 * 2);          // Wt1: KT=4, NT=16
  size_t WP4 = alloc((size_t)N_D * 8 *  8 * 32 * 16 * 2);          // Wt2: KT=8, NT=8
  // BN stats + folded scale/shift
  size_t SUM = alloc(4096 * sizeof(float));
  size_t SC1 = alloc(N_E * H1_F * 4), SH1 = alloc(N_E * H1_F * 4);
  size_t SC2 = alloc(N_E * H2_F * 4), SH2 = alloc(N_E * H2_F * 4);
  size_t SC3 = alloc(N_D * T1_F * 4), SH3 = alloc(N_D * T1_F * 4);
  size_t SC4 = alloc(N_D * T2_F * 4), SH4 = alloc(N_D * T2_F * 4);
  (void)ws_size; (void)in_sizes; (void)n_in; (void)out_size;

  float*  z1   = (float*)(ws + R1);
  float*  z2   = (float*)(ws + R2);
  float*  z3   = z1;
  float*  z4   = z2;
  float*  task = (float*)(ws + TK);
  float*  gts  = (float*)(ws + GT);
  __bf16* ApE  = (__bf16*)(ws + AP);
  __bf16* Wp1  = (__bf16*)(ws + WP1);
  __bf16* Wp2  = (__bf16*)(ws + WP2);
  __bf16* Wp3  = (__bf16*)(ws + WP3);
  __bf16* Wp4  = (__bf16*)(ws + WP4);
  float*  sums = (float*)(ws + SUM);
  float* sc1 = (float*)(ws + SC1); float* sh1 = (float*)(ws + SH1);
  float* sc2 = (float*)(ws + SC2); float* sh2 = (float*)(ws + SH2);
  float* sc3 = (float*)(ws + SC3); float* sh3 = (float*)(ws + SH3);
  float* sc4 = (float*)(ws + SC4); float* sh4 = (float*)(ws + SH4);

  const int MB = B_SZ / 128;   // 1024 M-blocks of 128 rows

  // ---- Pack weights + emb into WMMA fragment layout (bf16, zero-padded) ----
  pack_w<<<(N_E * 4 * 16 * 32 + 255) / 256, 256, 0, stream>>>(We1, Wp1, IN_F, H1_F, 4, 16, N_E);
  pack_w<<<(N_E * 8 *  8 * 32 + 255) / 256, 256, 0, stream>>>(We2, Wp2, H1_F, H2_F, 8,  8, N_E);
  pack_w<<<(N_D * 4 * 16 * 32 + 255) / 256, 256, 0, stream>>>(Wt1, Wp3, H2_F, T1_F, 4, 16, N_D);
  pack_w<<<(N_D * 8 *  8 * 32 + 255) / 256, 256, 0, stream>>>(Wt2, Wp4, T1_F, T2_F, 8,  8, N_D);
  pack_emb<<<((B_SZ / 16) * 4 * 32 + 255) / 256, 256, 0, stream>>>(emb, ApE);

  // ---- Expert layer 1: z1 = emb @ We1 + be1 (packed A, NT=16 -> grid.x=2) ----
  {
    int F = N_E * H1_F;
    zero_f32<<<(2 * F + 255) / 256, 256, 0, stream>>>(sums, 2 * F);
    gemm_t<IN_F, H1_F, 0, 1, 8><<<dim3(2, MB, N_E), 256, 0, stream>>>(
        nullptr, 0, 0, ApE, Wp1, be1, be1, be1, z1, sums, N_E);
    bn_finalize<<<(F + 255) / 256, 256, 0, stream>>>(sums, eg1, eb1, sc1, sh1, F);
  }

  // ---- Expert layer 2: z2 = relu(bn(z1)) @ We2 + be2 (NT=8 -> grid.x=1) ----
  {
    int F = N_E * H2_F;
    zero_f32<<<(2 * F + 255) / 256, 256, 0, stream>>>(sums, 2 * F);
    gemm_t<H1_F, H2_F, 1, 0, 8><<<dim3(1, MB, N_E), 256, 0, stream>>>(
        z1, (size_t)(N_E * H1_F), (size_t)H1_F, nullptr, Wp2, be2, sc1, sh1, z2, sums, N_E);
    bn_finalize<<<(F + 255) / 256, 256, 0, stream>>>(sums, eg2, eb2, sc2, sh2, F);
  }

  // ---- Gates + soft mixture (task padded to [B,D,128]) ----
  gates_kernel<<<(B_SZ * N_D + 255) / 256, 256, 0, stream>>>(emb, Wgp, bgp, gts);
  task_kernel<<<B_SZ, 128, 0, stream>>>(z2, sc2, sh2, gts, task);

  // ---- Tower layer 1: z3 = task @ Wt1 + bt1 (NT=16 -> grid.x=2) ----
  {
    int F = N_D * T1_F;
    zero_f32<<<(2 * F + 255) / 256, 256, 0, stream>>>(sums, 2 * F);
    gemm_t<H2_P, T1_F, 0, 0, 8><<<dim3(2, MB, N_D), 256, 0, stream>>>(
        task, (size_t)(N_D * H2_P), (size_t)H2_P, nullptr, Wp3, bt1, bt1, bt1, z3, sums, N_D);
    bn_finalize<<<(F + 255) / 256, 256, 0, stream>>>(sums, tg1, tb1, sc3, sh3, F);
  }

  // ---- Tower layer 2: z4 = relu(bn(z3)) @ Wt2 + bt2 (NT=8 -> grid.x=1) ----
  {
    int F = N_D * T2_F;
    zero_f32<<<(2 * F + 255) / 256, 256, 0, stream>>>(sums, 2 * F);
    gemm_t<T1_F, T2_F, 1, 0, 8><<<dim3(1, MB, N_D), 256, 0, stream>>>(
        z3, (size_t)(N_D * T1_F), (size_t)T1_F, nullptr, Wp4, bt2, sc3, sh3, z4, sums, N_D);
    bn_finalize<<<(F + 255) / 256, 256, 0, stream>>>(sums, tg2, tb2, sc4, sh4, F);
  }

  // ---- Tower head + sigmoid + domain gather ----
  final_kernel<<<B_SZ / 8, 256, 0, stream>>>(z4, sc4, sh4, Wt3, bt3, dom, out);
}
